// SlotAttention_9723805958436
// MI455X (gfx1250) — matrix operand order repaired and checked by reading, not compiled, not run on hardware
//
#include <hip/hip_runtime.h>

// Slot Attention on MI455X (gfx1250, wave32, WMMA bf16 16x16x32 + TDM).
// B=64, C=256, H=W=32 (N=1024), S=8 slots (padded to 16), D=256, 3 iters.
// All GEMMs run through v_wmma_f32_16x16x32_bf16 with f32 accumulation.
// x tile staging uses the Tensor Data Mover (tensor_load_to_lds) with LDS row
// padding (65-dword stride) so the LayerNorm reduction is bank-conflict-free.
// Workspace usage ~72 MB (k_bf 32MB + vT_bf 32MB + small buffers).

#define Bb 64
#define Cc 256
#define Nn 1024
#define Ss 8
#define Dd 256
#define D3 768
#define EPSc 1e-5f

typedef __attribute__((ext_vector_type(16))) __bf16 v16bf;
typedef __attribute__((ext_vector_type(8)))  float  v8f;
typedef unsigned int v4u __attribute__((ext_vector_type(4)));
typedef int          v8i __attribute__((ext_vector_type(8)));
typedef int          v4i __attribute__((ext_vector_type(4)));

union FragB { v16bf v; uint4 q[2]; };

__device__ __forceinline__ unsigned short f2bf(float f) {
  unsigned int u = __float_as_uint(f);
  u += 0x7FFFu + ((u >> 16) & 1u);      // round-to-nearest-even
  return (unsigned short)(u >> 16);
}

__device__ __forceinline__ v8f wmma_bf16(const FragB& a, const FragB& b, v8f acc) {
  // 8 args: (neg_a, A, neg_b, B, c_mod, C, reuse_a, reuse_b)
  return __builtin_amdgcn_wmma_f32_16x16x32_bf16(false, a.v, false, b.v,
                                                 (short)0, acc, false, false);
}

// ---------------------------------------------------------------- conversions
__global__ void k_cvt(const float* __restrict__ src, unsigned short* __restrict__ dst, int n) {
  int i = blockIdx.x * blockDim.x + threadIdx.x;
  if (i < n) dst[i] = f2bf(src[i]);
}

__global__ void k_init_slots(const float* __restrict__ mu, float* __restrict__ slots) {
  int i = blockIdx.x * blockDim.x + threadIdx.x;   // 131072 threads
  slots[i] = mu[i & (Ss * Dd - 1)];
}

// --------------------------------------------- fused TDM+LN -> K,V GEMM
// Grid: 1024 blocks (B x N/64), 256 threads (8 waves).
// LDS: xs[256][65] f32 (TDM-loaded x tile, channel-major, padded rows)
//    + abf[64][264] bf16 (LN'd, pixel-major A tile).
__global__ void k_ln_kv(const float* __restrict__ x,
                        const float* __restrict__ g, const float* __restrict__ be,
                        const unsigned short* __restrict__ wkb,
                        const unsigned short* __restrict__ wvb,
                        unsigned short* __restrict__ kbf,      // [B][N][D] bf16
                        unsigned short* __restrict__ vtbf) {   // [B][D][N] bf16
  extern __shared__ char smem[];
  float* xs = (float*)smem;                                    // [256][65]
  unsigned short* abf = (unsigned short*)(xs + 256 * 65);      // [64][264]
  const int tid = threadIdx.x, lane = tid & 31, wave = tid >> 5;
  const int b  = blockIdx.x >> 4;
  const int n0 = (blockIdx.x & 15) * 64;

#if __has_builtin(__builtin_amdgcn_tensor_load_to_lds) && __has_builtin(__builtin_amdgcn_s_wait_tensorcnt)
  // Tensor Data Mover: one DMA moves the whole 64KB tile x[b, 0:256, n0:n0+64]
  // into LDS with a 1-dword pad after every 64-dword row (pad_interval=5,
  // pad_amount=0) -> row stride 65 dwords, bank-conflict-free column reads.
  if (wave == 0) {
    const unsigned long long ga =
        (unsigned long long)(uintptr_t)(x + (size_t)b * Cc * Nn + n0);
    v4u g0;
    g0.x = 1u;                                       // count=1, no gather
    g0.y = 0u;                                       // lds_addr = 0 (xs base)
    g0.z = (unsigned int)(ga & 0xFFFFFFFFull);
    g0.w = (unsigned int)((ga >> 32) & 0x01FFFFFFull) | 0x80000000u;  // type=2
    v8i g1;
    g1[0] = (int)((2u << 16) | (1u << 20) | (5u << 22)); // data_size=4B, pad_en, pad/64dw
    g1[1] = (int)(1024u << 16);                      // tensor_dim0 = N = 1024
    g1[2] = (int)(256u << 16);                       // tensor_dim1 = C = 256
    g1[3] = (int)(64u << 16);                        // tile_dim0 = 64 pixels
    g1[4] = (int)256;                                // tile_dim1 = 256 channels
    g1[5] = (int)1024;                               // tensor_dim0_stride = N
    g1[6] = 0;
    g1[7] = 0;
    v4i gz4 = {0, 0, 0, 0};                          // 2-D tile: groups 2/3 unused
    v8i gz8 = {0, 0, 0, 0, 0, 0, 0, 0};
    // 6-arg toolchain form: (g0, g1, g2, g3, extra_group, cpol)
    __builtin_amdgcn_tensor_load_to_lds(g0, g1, gz4, gz4, gz8, 0);
    __builtin_amdgcn_s_wait_tensorcnt((short)0);
  }
#else
  for (int c = wave; c < Cc; c += 8) {
    const float2 t = *(const float2*)&x[(size_t)(b * Cc + c) * Nn + n0 + lane * 2];
    xs[c * 65 + lane * 2]     = t.x;
    xs[c * 65 + lane * 2 + 1] = t.y;
  }
#endif
  __syncthreads();

  // LayerNorm over C per pixel (xs is channel-major: xs[c*65 + pixel]);
  // wave handles 8 pixels; lanes stride channels -> 65-dword stride, no conflicts
  for (int pi = 0; pi < 8; ++pi) {
    const int p = wave * 8 + pi;
    float s = 0.f, s2 = 0.f;
    for (int c = lane; c < Cc; c += 32) { float v = xs[c * 65 + p]; s += v; s2 += v * v; }
    for (int m = 16; m >= 1; m >>= 1) { s += __shfl_xor(s, m); s2 += __shfl_xor(s2, m); }
    const float mu = s * (1.f / Cc);
    const float rs = rsqrtf(s2 * (1.f / Cc) - mu * mu + EPSc);
    for (int c = lane; c < Cc; c += 32)
      abf[p * 264 + c] = f2bf((xs[c * 65 + p] - mu) * rs * g[c] + be[c]);
  }
  __syncthreads();

  // GEMM: 32 output columns (16 d-tiles for K + 16 for V), 4 per wave.
  // Per column: load the 8 B-fragments (weight rows) ONCE into registers
  // (64 VGPRs) and reuse them across all 4 row-tile accumulation chains ->
  // 4x less weight traffic from L2 without keeping A-fragments live.
  const int ml = lane & 15, hi = lane >> 4;
  for (int j = 0; j < 4; ++j) {
    // Block LICM from hoisting the (j-invariant) LDS A-fragment loads out of
    // this loop — that caused scratch spills. Zero-instruction compiler fence.
    asm volatile("" ::: "memory");
    const int col = wave + 8 * j;          // 0..31
    const int is_v = col >> 4;
    const int nt = col & 15;
    const unsigned short* Wm = is_v ? wvb : wkb;
    FragB bfr[8];
#pragma unroll
    for (int kk = 0; kk < 8; ++kk) {
      const unsigned short* wr = &Wm[(nt * 16 + ml) * Cc + kk * 32 + hi * 16];
      bfr[kk].q[0] = *(const uint4*)wr;
      bfr[kk].q[1] = *(const uint4*)(wr + 8);
    }
#pragma unroll
    for (int mt = 0; mt < 4; ++mt) {
      v8f acc = {0.f,0.f,0.f,0.f,0.f,0.f,0.f,0.f};
#pragma unroll
      for (int kk = 0; kk < 8; ++kk) {
        FragB a;
        const unsigned short* ar = &abf[(mt * 16 + ml) * 264 + kk * 32 + hi * 16];
        a.q[0] = *(const uint4*)ar;
        a.q[1] = *(const uint4*)(ar + 8);
        acc = wmma_bf16(a, bfr[kk], acc);
      }
      const int d = nt * 16 + ml;
      if (!is_v) {                     // k stored [n][d]
#pragma unroll
        for (int r = 0; r < 8; ++r) {
          const int n = n0 + mt * 16 + r + 8 * hi;
          kbf[(size_t)(b * Nn + n) * Dd + d] = f2bf(acc[r]);
        }
      } else {                         // v stored transposed [d][n] (K-major)
        uint4 pk;
        pk.x = (unsigned)f2bf(acc[0]) | ((unsigned)f2bf(acc[1]) << 16);
        pk.y = (unsigned)f2bf(acc[2]) | ((unsigned)f2bf(acc[3]) << 16);
        pk.z = (unsigned)f2bf(acc[4]) | ((unsigned)f2bf(acc[5]) << 16);
        pk.w = (unsigned)f2bf(acc[6]) | ((unsigned)f2bf(acc[7]) << 16);
        *(uint4*)&vtbf[(size_t)(b * Dd + d) * Nn + n0 + mt * 16 + 8 * hi] = pk;
      }
    }
  }
}

// -------------------------------------------------- q = LN(slots) @ wq^T * scale
__global__ void k_qproj(const float* __restrict__ slots,
                        const float* __restrict__ gs, const float* __restrict__ bs,
                        const unsigned short* __restrict__ wqb,
                        unsigned short* __restrict__ qpad) {
  __shared__ __align__(16) unsigned short sb[16 * 264];
  const int tid = threadIdx.x, lane = tid & 31, wave = tid >> 5;
  const int t = blockIdx.x;
  for (int i = 0; i < 2; ++i) {
    const int row = wave * 2 + i;
    const int gr  = t * 16 + row;
    float s = 0.f, s2 = 0.f;
    for (int c = lane; c < Dd; c += 32) { float v = slots[gr * Dd + c]; s += v; s2 += v * v; }
    for (int m = 16; m >= 1; m >>= 1) { s += __shfl_xor(s, m); s2 += __shfl_xor(s2, m); }
    const float mu = s / (float)Dd;
    const float rs = rsqrtf(s2 / (float)Dd - mu * mu + EPSc);
    for (int c = lane; c < Dd; c += 32)
      sb[row * 264 + c] = f2bf((slots[gr * Dd + c] - mu) * rs * gs[c] + bs[c]);
  }
  // zero padding rows s=8..15 of q_pad for batches 2t, 2t+1
  for (int idx = tid; idx < 16 * Dd; idx += 256) {
    const int rr = idx >> 8, c = idx & 255;
    const int bq = 2 * t + (rr >> 3), sp = 8 + (rr & 7);
    qpad[(bq * 16 + sp) * Dd + c] = 0;
  }
  __syncthreads();
  const int ml = lane & 15, hi = lane >> 4;
  for (int j = 0; j < 2; ++j) {
    const int nt = wave * 2 + j;
    v8f acc = {0.f,0.f,0.f,0.f,0.f,0.f,0.f,0.f};
#pragma unroll
    for (int k0 = 0; k0 < Dd; k0 += 32) {
      FragB a, bb;
      const unsigned short* ar = &sb[ml * 264 + k0 + hi * 16];
      a.q[0] = *(const uint4*)ar; a.q[1] = *(const uint4*)(ar + 8);
      const unsigned short* wr = &wqb[(nt * 16 + ml) * Dd + k0 + hi * 16];
      bb.q[0] = *(const uint4*)wr; bb.q[1] = *(const uint4*)(wr + 8);
      acc = wmma_bf16(a, bb, acc);
    }
    const int d = nt * 16 + ml;
#pragma unroll
    for (int r = 0; r < 8; ++r) {
      const int gr = t * 16 + r + 8 * hi;
      const int bq = gr >> 3, sp = gr & 7;
      qpad[(bq * 16 + sp) * Dd + d] = f2bf(acc[r] * 0.0625f);  // fold D^-0.5 scale
    }
  }
}

// ---------------------------------------- logits + softmax(S) -> attn, attn^T
__global__ void k_attn(const unsigned short* __restrict__ kbf,
                       const unsigned short* __restrict__ qpad,
                       float* __restrict__ attn_out,            // [B][N][S] f32 (d_out)
                       unsigned short* __restrict__ attnT) {    // [B][16][N] bf16
  const int tid = threadIdx.x, lane = tid & 31, wave = tid >> 5;
  const int b = blockIdx.y;
  const int nbase = blockIdx.x * 128 + wave * 16;
  const int ml = lane & 15, hi = lane >> 4;
  v8f acc = {0.f,0.f,0.f,0.f,0.f,0.f,0.f,0.f};
#pragma unroll
  for (int k0 = 0; k0 < Dd; k0 += 32) {
    FragB a, bb;
    const unsigned short* ar = &kbf[(size_t)(b * Nn + nbase + ml) * Dd + k0 + hi * 16];
    a.q[0] = *(const uint4*)ar; a.q[1] = *(const uint4*)(ar + 8);
    const unsigned short* qr = &qpad[(b * 16 + ml) * Dd + k0 + hi * 16];
    bb.q[0] = *(const uint4*)qr; bb.q[1] = *(const uint4*)(qr + 8);
    acc = wmma_bf16(a, bb, acc);
  }
  // softmax over the 8 slot columns; cols 8..15 are padding (masked to -inf)
  float av[8];
#pragma unroll
  for (int r = 0; r < 8; ++r) {
    float v = (ml < 8) ? acc[r] : -3.0e38f;
    float mx = v;
    mx = fmaxf(mx, __shfl_xor(mx, 1));
    mx = fmaxf(mx, __shfl_xor(mx, 2));
    mx = fmaxf(mx, __shfl_xor(mx, 4));
    mx = fmaxf(mx, __shfl_xor(mx, 8));
    float e = (ml < 8) ? __expf(v - mx) : 0.f;
    float sm = e;
    sm += __shfl_xor(sm, 1);
    sm += __shfl_xor(sm, 2);
    sm += __shfl_xor(sm, 4);
    sm += __shfl_xor(sm, 8);
    av[r] = e / sm;
  }
  if (ml < 8) {
#pragma unroll
    for (int r = 0; r < 8; ++r) {
      const int n = nbase + r + 8 * hi;
      attn_out[(size_t)(b * Nn + n) * Ss + ml] = av[r];
    }
  }
  // attn^T (K-major for the updates GEMM); pad columns store exact zeros
  uint4 pk;
  pk.x = (unsigned)f2bf(av[0]) | ((unsigned)f2bf(av[1]) << 16);
  pk.y = (unsigned)f2bf(av[2]) | ((unsigned)f2bf(av[3]) << 16);
  pk.z = (unsigned)f2bf(av[4]) | ((unsigned)f2bf(av[5]) << 16);
  pk.w = (unsigned)f2bf(av[6]) | ((unsigned)f2bf(av[7]) << 16);
  *(uint4*)&attnT[(size_t)(b * 16 + ml) * Nn + nbase + 8 * hi] = pk;
}

// --------------------------------------------- updates = attn^T @ V  (K = 1024)
__global__ void k_updates(const unsigned short* __restrict__ attnT,
                          const unsigned short* __restrict__ vtbf,
                          float* __restrict__ upd) {            // [B*S][D] f32
  const int tid = threadIdx.x, lane = tid & 31, wave = tid >> 5;
  const int b  = blockIdx.y;
  const int nt = blockIdx.x * 8 + wave;
  const int ml = lane & 15, hi = lane >> 4;
  const int d  = nt * 16 + ml;
  v8f acc = {0.f,0.f,0.f,0.f,0.f,0.f,0.f,0.f};
  const unsigned short* arow = &attnT[(size_t)(b * 16 + ml) * Nn];
  const unsigned short* vrow = &vtbf[(size_t)(b * Dd + d) * Nn];
  for (int k0 = 0; k0 < Nn; k0 += 32) {
    FragB a, bb;
    a.q[0]  = *(const uint4*)(arow + k0 + hi * 16);
    a.q[1]  = *(const uint4*)(arow + k0 + hi * 16 + 8);
    bb.q[0] = *(const uint4*)(vrow + k0 + hi * 16);
    bb.q[1] = *(const uint4*)(vrow + k0 + hi * 16 + 8);
    acc = wmma_bf16(a, bb, acc);
  }
  if (hi == 0) {                      // rows 0..7 are the 8 real slots
#pragma unroll
    for (int r = 0; r < 8; ++r)
      upd[(size_t)(b * Ss + r) * Dd + d] = acc[r];
  }
}

// ------------------------------ fused GRUCell + LN + MLP residual (per 16 rows)
__global__ void k_gru_mlp(const float* __restrict__ upd,
                          float* __restrict__ slots,
                          const unsigned short* __restrict__ wihb,
                          const unsigned short* __restrict__ whhb,
                          const float* __restrict__ bih, const float* __restrict__ bhh,
                          const float* __restrict__ gm,  const float* __restrict__ bm,
                          const unsigned short* __restrict__ w1b, const float* __restrict__ b1,
                          const unsigned short* __restrict__ w2b, const float* __restrict__ b2) {
  extern __shared__ char smem[];
  unsigned short* aU = (unsigned short*)smem;          // [16][264] bf16 updates
  unsigned short* aH = aU + 16 * 264;                  // [16][264] bf16 slots
  float* Rg = (float*)(aH + 16 * 264);                 // [16][256]
  float* Zg = Rg + 16 * 256;
  float* NN = Zg + 16 * 256;
  float* HN = NN + 16 * 256;
  float* hp = HN + 16 * 256;
  unsigned short* aL = (unsigned short*)(hp + 16 * 256);
  unsigned short* aM = aL + 16 * 264;
  const int tid = threadIdx.x, lane = tid & 31, wave = tid >> 5;
  const int g0 = blockIdx.x * 16;

  for (int idx = tid; idx < 16 * Dd; idx += 256) {
    const int row = idx >> 8, c = idx & 255;
    aU[row * 264 + c] = f2bf(upd[(g0 + row) * Dd + c]);
    aH[row * 264 + c] = f2bf(slots[(g0 + row) * Dd + c]);
  }
  __syncthreads();
  const int ml = lane & 15, hi = lane >> 4;

  // gi = upd@wih^T, gh = h@whh^T over 3D=768 outputs: 48 col tiles, 6 per wave
  for (int ot = wave; ot < 48; ot += 8) {
    const int o0 = ot * 16;
    v8f ai = {0.f,0.f,0.f,0.f,0.f,0.f,0.f,0.f};
    v8f ah = {0.f,0.f,0.f,0.f,0.f,0.f,0.f,0.f};
#pragma unroll
    for (int k0 = 0; k0 < Dd; k0 += 32) {
      FragB a, bb;
      const unsigned short* ar = &aU[ml * 264 + k0 + hi * 16];
      a.q[0] = *(const uint4*)ar; a.q[1] = *(const uint4*)(ar + 8);
      const unsigned short* wr = &wihb[(o0 + ml) * Dd + k0 + hi * 16];
      bb.q[0] = *(const uint4*)wr; bb.q[1] = *(const uint4*)(wr + 8);
      ai = wmma_bf16(a, bb, ai);
      const unsigned short* hr = &aH[ml * 264 + k0 + hi * 16];
      a.q[0] = *(const uint4*)hr; a.q[1] = *(const uint4*)(hr + 8);
      const unsigned short* wr2 = &whhb[(o0 + ml) * Dd + k0 + hi * 16];
      bb.q[0] = *(const uint4*)wr2; bb.q[1] = *(const uint4*)(wr2 + 8);
      ah = wmma_bf16(a, bb, ah);
    }
    const int o = o0 + ml;
    const float bi = bih[o], bh = bhh[o];
#pragma unroll
    for (int r = 0; r < 8; ++r) {
      const int row = r + 8 * hi;
      if (o < 256)      Rg[row * 256 + o]         = ai[r] + ah[r] + bi + bh;
      else if (o < 512) Zg[row * 256 + (o - 256)] = ai[r] + ah[r] + bi + bh;
      else { NN[row * 256 + (o - 512)] = ai[r] + bi;
             HN[row * 256 + (o - 512)] = ah[r] + bh; }
    }
  }
  __syncthreads();

  // gates -> h'
  for (int idx = tid; idx < 16 * Dd; idx += 256) {
    const int row = idx >> 8, c = idx & 255;
    const float r_ = 1.f / (1.f + __expf(-Rg[row * 256 + c]));
    const float z_ = 1.f / (1.f + __expf(-Zg[row * 256 + c]));
    const float n_ = tanhf(NN[row * 256 + c] + r_ * HN[row * 256 + c]);
    const float h_ = slots[(g0 + row) * Dd + c];
    hp[row * 256 + c] = (1.f - z_) * n_ + z_ * h_;
  }
  __syncthreads();

  // LN(h') -> aL
  for (int i = 0; i < 2; ++i) {
    const int row = wave * 2 + i;
    float s = 0.f, s2 = 0.f;
    for (int c = lane; c < Dd; c += 32) { float v = hp[row * 256 + c]; s += v; s2 += v * v; }
    for (int m = 16; m >= 1; m >>= 1) { s += __shfl_xor(s, m); s2 += __shfl_xor(s2, m); }
    const float mu = s / (float)Dd;
    const float rs = rsqrtf(s2 / (float)Dd - mu * mu + EPSc);
    for (int c = lane; c < Dd; c += 32)
      aL[row * 264 + c] = f2bf((hp[row * 256 + c] - mu) * rs * gm[c] + bm[c]);
  }
  __syncthreads();

  // hidden = relu(aL@w1^T + b1) -> aM
  for (int j = 0; j < 2; ++j) {
    const int nt = wave * 2 + j;
    v8f acc = {0.f,0.f,0.f,0.f,0.f,0.f,0.f,0.f};
#pragma unroll
    for (int k0 = 0; k0 < Dd; k0 += 32) {
      FragB a, bb;
      const unsigned short* ar = &aL[ml * 264 + k0 + hi * 16];
      a.q[0] = *(const uint4*)ar; a.q[1] = *(const uint4*)(ar + 8);
      const unsigned short* wr = &w1b[(nt * 16 + ml) * Dd + k0 + hi * 16];
      bb.q[0] = *(const uint4*)wr; bb.q[1] = *(const uint4*)(wr + 8);
      acc = wmma_bf16(a, bb, acc);
    }
    const int d = nt * 16 + ml;
    const float bias = b1[d];
#pragma unroll
    for (int r = 0; r < 8; ++r) {
      const float v = acc[r] + bias;
      aM[(r + 8 * hi) * 264 + d] = f2bf(v > 0.f ? v : 0.f);
    }
  }
  __syncthreads();

  // slots = h' + (aM@w2^T + b2)
  for (int j = 0; j < 2; ++j) {
    const int nt = wave * 2 + j;
    v8f acc = {0.f,0.f,0.f,0.f,0.f,0.f,0.f,0.f};
#pragma unroll
    for (int k0 = 0; k0 < Dd; k0 += 32) {
      FragB a, bb;
      const unsigned short* ar = &aM[ml * 264 + k0 + hi * 16];
      a.q[0] = *(const uint4*)ar; a.q[1] = *(const uint4*)(ar + 8);
      const unsigned short* wr = &w2b[(nt * 16 + ml) * Dd + k0 + hi * 16];
      bb.q[0] = *(const uint4*)wr; bb.q[1] = *(const uint4*)(wr + 8);
      acc = wmma_bf16(a, bb, acc);
    }
    const int d = nt * 16 + ml;
    const float bias = b2[d];
#pragma unroll
    for (int r = 0; r < 8; ++r) {
      const int row = r + 8 * hi;
      slots[(g0 + row) * Dd + d] = hp[row * 256 + d] + acc[r] + bias;
    }
  }
}

// --------------------------------------------------------------------- driver
extern "C" void kernel_launch(void* const* d_in, const int* in_sizes, int n_in,
                              void* d_out, int out_size, void* d_ws, size_t ws_size,
                              hipStream_t stream) {
  (void)in_sizes; (void)n_in; (void)out_size; (void)ws_size;
  const float* x        = (const float*)d_in[0];
  const float* slots_mu = (const float*)d_in[1];
  const float* ln_in_g  = (const float*)d_in[2];
  const float* ln_in_b  = (const float*)d_in[3];
  const float* wk       = (const float*)d_in[4];
  const float* wv       = (const float*)d_in[5];
  const float* ln_s_g   = (const float*)d_in[6];
  const float* ln_s_b   = (const float*)d_in[7];
  const float* wq       = (const float*)d_in[8];
  const float* wih      = (const float*)d_in[9];
  const float* whh      = (const float*)d_in[10];
  const float* bih      = (const float*)d_in[11];
  const float* bhh      = (const float*)d_in[12];
  const float* ln_m_g   = (const float*)d_in[13];
  const float* ln_m_b   = (const float*)d_in[14];
  const float* w1       = (const float*)d_in[15];
  const float* b1       = (const float*)d_in[16];
  const float* w2       = (const float*)d_in[17];
  const float* b2       = (const float*)d_in[18];

  char* ws = (char*)d_ws;
  size_t off = 0;
  auto carve = [&](size_t bytes) -> void* {
    void* p = ws + off; off += (bytes + 255) & ~(size_t)255; return p;
  };
  unsigned short* wkb   = (unsigned short*)carve((size_t)Dd * Cc * 2);
  unsigned short* wvb   = (unsigned short*)carve((size_t)Dd * Cc * 2);
  unsigned short* wqb   = (unsigned short*)carve((size_t)Dd * Dd * 2);
  unsigned short* wihb  = (unsigned short*)carve((size_t)D3 * Dd * 2);
  unsigned short* whhb  = (unsigned short*)carve((size_t)D3 * Dd * 2);
  unsigned short* w1b   = (unsigned short*)carve((size_t)Dd * Dd * 2);
  unsigned short* w2b   = (unsigned short*)carve((size_t)Dd * Dd * 2);
  float*          slots = (float*)carve((size_t)Bb * Ss * Dd * 4);
  unsigned short* qpad  = (unsigned short*)carve((size_t)Bb * 16 * Dd * 2);
  float*          updf  = (float*)carve((size_t)Bb * Ss * Dd * 4);
  unsigned short* attnT = (unsigned short*)carve((size_t)Bb * 16 * Nn * 2);
  unsigned short* kbf   = (unsigned short*)carve((size_t)Bb * Nn * Dd * 2);  // 32 MB
  unsigned short* vtbf  = (unsigned short*)carve((size_t)Bb * Dd * Nn * 2);  // 32 MB

  k_cvt<<<Dd * Cc / 256, 256, 0, stream>>>(wk,  wkb,  Dd * Cc);
  k_cvt<<<Dd * Cc / 256, 256, 0, stream>>>(wv,  wvb,  Dd * Cc);
  k_cvt<<<Dd * Dd / 256, 256, 0, stream>>>(wq,  wqb,  Dd * Dd);
  k_cvt<<<D3 * Dd / 256, 256, 0, stream>>>(wih, wihb, D3 * Dd);
  k_cvt<<<D3 * Dd / 256, 256, 0, stream>>>(whh, whhb, D3 * Dd);
  k_cvt<<<Dd * Dd / 256, 256, 0, stream>>>(w1,  w1b,  Dd * Dd);
  k_cvt<<<Dd * Dd / 256, 256, 0, stream>>>(w2,  w2b,  Dd * Dd);
  k_init_slots<<<Bb * Ss * Dd / 256, 256, 0, stream>>>(slots_mu, slots);

  const size_t smemA = 256 * 65 * sizeof(float) + 64 * 264 * sizeof(unsigned short);
  k_ln_kv<<<Bb * 16, 256, smemA, stream>>>(x, ln_in_g, ln_in_b, wkb, wvb, kbf, vtbf);

  float* attn_out = (float*)d_out + (size_t)Bb * Ss * Dd;  // slots first, attn second
  const size_t smemG = 2 * (16 * 264 * 2) + 5 * (16 * 256 * 4) + 2 * (16 * 264 * 2);
  for (int it = 0; it < 3; ++it) {
    k_qproj<<<32, 256, 0, stream>>>(slots, ln_s_g, ln_s_b, wqb, qpad);
    k_attn<<<dim3(8, Bb), 256, 0, stream>>>(kbf, qpad, attn_out, attnT);
    k_updates<<<dim3(2, Bb), 256, 0, stream>>>(attnT, vtbf, updf);
    k_gru_mlp<<<32, 256, smemG, stream>>>(updf, slots, wihb, whhb, bih, bhh,
                                          ln_m_g, ln_m_b, w1b, b1, w2b, b2);
  }
  (void)hipMemcpyAsync(d_out, slots, (size_t)Bb * Ss * Dd * sizeof(float),
                       hipMemcpyDeviceToDevice, stream);
}
